// SoftDecisionTree_2430951489980
// MI455X (gfx1250) — matrix operand order repaired and checked by reading, compiled, not run on hardware
//
#include <hip/hip_runtime.h>
#include <hip/hip_bf16.h>

typedef __attribute__((ext_vector_type(16))) __bf16 v16bf;
typedef __attribute__((ext_vector_type(8)))  __bf16 v8bf;
typedef __attribute__((ext_vector_type(8)))  float  v8f;
typedef __attribute__((ext_vector_type(4)))  float  v4f;

#define BATCH     8192
#define INPUT_DIM 2048
#define N_INT     1023
#define N_PAD     1024
#define PHI_N     24
#define TILE_M    128
#define TILE_N    128
#define TILE_K    32
#define K_BLOCKS  (INPUT_DIM / TILE_K)      // 64
#define M_GROUPS  (BATCH / 16)              // 512
#define N_GROUPS  (N_PAD / 16)              // 64

// Detect gfx1250 async global->LDS copy support (device pass only; host pass
// and older toolchains fall back to a synchronous copy).
#if __has_builtin(__builtin_amdgcn_global_load_async_to_lds_b128)
#define SDT_ASYNC_COPY 1
// Exact parameter types per hipcc diagnostic: (V4i addrspace(1)*, V4i addrspace(3)*, Ii, Ii)
typedef int sdt_v4i __attribute__((vector_size(16)));
typedef __attribute__((address_space(1))) sdt_v4i* sdt_gptr;
typedef __attribute__((address_space(3))) sdt_v4i* sdt_lptr;
#else
#define SDT_ASYNC_COPY 0
#endif

// Convert 8 consecutive fp32 -> 8 bf16.
__device__ __forceinline__ v8bf cvt8(const float* __restrict__ src) {
    v4f f0 = *(const v4f*)(src);
    v4f f1 = *(const v4f*)(src + 4);
    v8bf h;
    h[0] = (__bf16)f0[0]; h[1] = (__bf16)f0[1]; h[2] = (__bf16)f0[2]; h[3] = (__bf16)f0[3];
    h[4] = (__bf16)f1[0]; h[5] = (__bf16)f1[1]; h[6] = (__bf16)f1[2]; h[7] = (__bf16)f1[3];
    return h;
}

// ---------------------------------------------------------------------------
// Pre-pass: convert xs / W to bf16, stored as the exact WMMA-fragment LDS
// image so the GEMM stages tiles with identity byte copies.
//
// A image: [kBlock][mGroup][lane 0..31][16 halves]  (1KB per 16x32 tile)
//   A frag (16x32 MxK): lane<16 : row=lane,    halves = K[0..7],K[16..23]
//                       lane>=16: row=lane-16, halves = K[8..15],K[24..31]
// B image: [kBlock][nGroup][lane 0..31][16 halves]
//   B frag (32x16 KxN): lane<16 : col=lane,    halves = K[0..7],K[16..23]
//                       lane>=16: col=lane-16, halves = K[8..15],K[24..31]
// ---------------------------------------------------------------------------
__global__ __launch_bounds__(256)
void sdt_swizzle_A(const float* __restrict__ xs, __bf16* __restrict__ Asw)
{
    const int idx  = (int)blockIdx.x * 256 + (int)threadIdx.x;  // one K-octet each
    const int m    = idx >> 8;            // 256 octets per row
    const int k    = (idx & 255) * 8;
    v8bf h = cvt8(xs + (size_t)m * INPUT_DIM + k);
    const int kBlock   = k >> 5;
    const int o        = (k & 31) >> 3;   // octet within K-block
    const int lane     = (m & 15) | ((o & 1) << 4);
    const int halfBase = (o >> 1) * 8;
    const size_t off = ((size_t)kBlock * M_GROUPS + (m >> 4)) * 512 + lane * 16 + halfBase;
    *(v8bf*)(Asw + off) = h;
}

__global__ __launch_bounds__(256)
void sdt_swizzle_B(const float* __restrict__ W, __bf16* __restrict__ Bsw)
{
    const int idx = (int)blockIdx.x * 256 + (int)threadIdx.x;
    const int n   = idx >> 8;             // padded output column 0..1023
    const int k   = (idx & 255) * 8;
    int wrow = n; if (wrow > N_INT - 1) wrow = N_INT - 1;   // pad row 1023
    v8bf h = cvt8(W + (size_t)wrow * INPUT_DIM + k);
    const int kBlock   = k >> 5;
    const int o        = (k & 31) >> 3;
    const int lane     = (n & 15) | ((o & 1) << 4);
    const int halfBase = (o >> 1) * 8;
    const size_t off = ((size_t)kBlock * N_GROUPS + (n >> 4)) * 512 + lane * 16 + halfBase;
    *(v8bf*)(Bsw + off) = h;
}

// ---------------------------------------------------------------------------
// Kernel 1: probs = sigmoid(A @ B + b) with bf16 WMMA, f32 accumulate.
// 256 threads (8 waves), block tile 128x128, wave tile 32x64.
// Double-buffered LDS; staging is an identity copy of two contiguous 8KB
// regions, issued via GLOBAL_LOAD_ASYNC_TO_LDS_B128 when available and
// overlapped with the WMMAs of the current K-step.
// ---------------------------------------------------------------------------
__device__ __forceinline__ void sdt_stage(const __bf16* __restrict__ gA,
                                          const __bf16* __restrict__ gB,
                                          __bf16* ldsbuf, int wave, int lane)
{
    // waves 0-3 copy the 8KB A panel, waves 4-7 the 8KB B panel: 2KB per wave,
    // 4 chunks of 512B (16B per lane per chunk).
    const __bf16* src = (wave < 4) ? (gA + wave * 1024) : (gB + (wave - 4) * 1024);
    __bf16*       dst = (wave < 4) ? (ldsbuf + wave * 1024)
                                   : (ldsbuf + 4096 + (wave - 4) * 1024);
#pragma unroll
    for (int i = 0; i < 4; ++i) {
        const __bf16* s = src + i * 256 + lane * 8;
        __bf16*       d = dst + i * 256 + lane * 8;
#if SDT_ASYNC_COPY
        __builtin_amdgcn_global_load_async_to_lds_b128(
            (sdt_gptr)(s), (sdt_lptr)(d), /*offset=*/0, /*cpol=*/0);
#else
        *(v8bf*)d = *(const v8bf*)s;
#endif
    }
}

__global__ __launch_bounds__(256)
void sdt_gemm_sigmoid(const __bf16* __restrict__ Asw,
                      const __bf16* __restrict__ Bsw,
                      const float* __restrict__ b,
                      float* __restrict__ probs)
{
    __shared__ alignas(128) __bf16 lds[2 * 8192];   // 2 x (8KB A + 8KB B)

    const int tid  = threadIdx.x;
    const int lane = tid & 31;
    const int wave = tid >> 5;
    const int wm   = wave >> 1;        // 0..3 : 32-row slab
    const int wn   = wave & 1;         // 0..1 : 64-col slab

    const int mBase = (int)(blockIdx.x >> 3) * TILE_M;
    const int nBase = (int)(blockIdx.x & 7) * TILE_N;
    const int mG0   = mBase >> 4;      // first of 8 A groups
    const int nG0   = nBase >> 4;      // first of 8 B groups

    v8f acc[2][4];
#pragma unroll
    for (int i = 0; i < 2; ++i)
#pragma unroll
        for (int t = 0; t < 4; ++t)
#pragma unroll
            for (int e = 0; e < 8; ++e)
                acc[i][t][e] = 0.0f;

    // Prologue: stage K-step 0 into buffer 0.
    sdt_stage(Asw + ((size_t)0 * M_GROUPS + mG0) * 512,
              Bsw + ((size_t)0 * N_GROUPS + nG0) * 512,
              lds, wave, lane);

    for (int i = 0; i < K_BLOCKS; ++i) {
#if SDT_ASYNC_COPY
#if __has_builtin(__builtin_amdgcn_s_wait_asynccnt)
        __builtin_amdgcn_s_wait_asynccnt(0);
#else
        asm volatile("s_wait_asynccnt 0x0" ::: "memory");
#endif
#endif
        __syncthreads();   // buffer i&1 fully populated; buffer (i+1)&1 free

        if (i + 1 < K_BLOCKS)
            sdt_stage(Asw + ((size_t)(i + 1) * M_GROUPS + mG0) * 512,
                      Bsw + ((size_t)(i + 1) * N_GROUPS + nG0) * 512,
                      lds + ((i + 1) & 1) * 8192, wave, lane);

        const __bf16* bufA = lds + (i & 1) * 8192;
        const __bf16* bufB = bufA + 4096;

        const v16bf a0 = *(const v16bf*)(bufA + (wm * 2 + 0) * 512 + lane * 16);
        const v16bf a1 = *(const v16bf*)(bufA + (wm * 2 + 1) * 512 + lane * 16);
        v16bf bf[4];
#pragma unroll
        for (int t = 0; t < 4; ++t)
            bf[t] = *(const v16bf*)(bufB + (wn * 4 + t) * 512 + lane * 16);
#pragma unroll
        for (int t = 0; t < 4; ++t) {
            acc[0][t] = __builtin_amdgcn_wmma_f32_16x16x32_bf16(
                false, a0, false, bf[t], (short)0, acc[0][t], false, false);
            acc[1][t] = __builtin_amdgcn_wmma_f32_16x16x32_bf16(
                false, a1, false, bf[t], (short)0, acc[1][t], false, false);
        }
    }

    // Epilogue: bias + sigmoid. C/D layout: lane0-15: N=lane, VGPR r -> M=r;
    // lane16-31: N=lane-16, VGPR r -> M=r+8.
    const int nLocal = lane & 15;
    const int mHi    = (lane >> 4) * 8;
#pragma unroll
    for (int t = 0; t < 4; ++t) {
        const int col = nBase + wn * 64 + t * 16 + nLocal;
        int bi = col; if (bi > N_INT - 1) bi = N_INT - 1;
        const float bias = b[bi];
#pragma unroll
        for (int i = 0; i < 2; ++i) {
            const int mRow = mBase + wm * 32 + i * 16 + mHi;
#pragma unroll
            for (int r2 = 0; r2 < 8; ++r2) {
                const float v = acc[i][t][r2] + bias;
                const float p = 1.0f / (1.0f + __expf(-v));
                probs[(size_t)(mRow + r2) * N_PAD + col] = p;
            }
        }
    }
}

// ---------------------------------------------------------------------------
// Kernel 2: path probabilities + out = pp @ phi. One wave per 4 batch rows;
// lane L owns leaves [32L, 32L+32): 5-level prefix product (lane-only) times
// an in-register 5-level subtree expansion, outer-product FMA against phi
// rows [32L,32L+32) shared across the 4 rows, wave32 shfl_xor reduction.
// ---------------------------------------------------------------------------
__global__ __launch_bounds__(256)
void sdt_tree_phi(const float* __restrict__ probs,
                  const float* __restrict__ phi,
                  float* __restrict__ out)
{
    const int lane    = threadIdx.x & 31;
    const int wave    = threadIdx.x >> 5;
    const int rowBase = (int)blockIdx.x * 32 + wave * 4;

    const int n1 = 1 + (lane >> 4);
    const int n2 = 3 + (lane >> 3);
    const int n3 = 7 + (lane >> 2);
    const int n4 = 15 + (lane >> 1);
    const int b0 = (lane >> 4) & 1, b1 = (lane >> 3) & 1, b2 = (lane >> 2) & 1,
              b3 = (lane >> 1) & 1, b4 = lane & 1;

    float pp[4][32];
#pragma unroll
    for (int r = 0; r < 4; ++r) {
        const float* pr = probs + (size_t)(rowBase + r) * N_PAD;
        const float p0 = pr[0], p1 = pr[n1], p2 = pr[n2], p3 = pr[n3], p4 = pr[n4];
        float pre = (b0 ? p0 : 1.0f - p0);
        pre *= (b1 ? p1 : 1.0f - p1);
        pre *= (b2 ? p2 : 1.0f - p2);
        pre *= (b3 ? p3 : 1.0f - p3);
        pre *= (b4 ? p4 : 1.0f - p4);

        const float p5 = pr[31 + lane];
        float q1[2];
        q1[0] = pre * (1.0f - p5); q1[1] = pre * p5;
        float q2[4];
#pragma unroll
        for (int t = 0; t < 2; ++t) {
            const float p = pr[63 + 2 * lane + t];
            q2[2 * t] = q1[t] * (1.0f - p); q2[2 * t + 1] = q1[t] * p;
        }
        float q3[8];
#pragma unroll
        for (int t = 0; t < 4; ++t) {
            const float p = pr[127 + 4 * lane + t];
            q3[2 * t] = q2[t] * (1.0f - p); q3[2 * t + 1] = q2[t] * p;
        }
        float q4[16];
#pragma unroll
        for (int t = 0; t < 8; ++t) {
            const float p = pr[255 + 8 * lane + t];
            q4[2 * t] = q3[t] * (1.0f - p); q4[2 * t + 1] = q3[t] * p;
        }
#pragma unroll
        for (int t = 0; t < 16; ++t) {
            const float p = pr[511 + 16 * lane + t];
            pp[r][2 * t] = q4[t] * (1.0f - p); pp[r][2 * t + 1] = q4[t] * p;
        }
    }

    float accv[4][PHI_N];
#pragma unroll
    for (int r = 0; r < 4; ++r)
#pragma unroll
        for (int c = 0; c < PHI_N; ++c)
            accv[r][c] = 0.0f;

    const float* phiL = phi + (size_t)lane * 32 * PHI_N;
#pragma unroll
    for (int j = 0; j < 32; ++j) {
        v4f ph[6];
#pragma unroll
        for (int q = 0; q < 6; ++q)
            ph[q] = *(const v4f*)(phiL + j * PHI_N + q * 4);
#pragma unroll
        for (int r = 0; r < 4; ++r) {
            const float w = pp[r][j];
#pragma unroll
            for (int c = 0; c < PHI_N; ++c)
                accv[r][c] = fmaf(w, ph[c >> 2][c & 3], accv[r][c]);
        }
    }

#pragma unroll
    for (int r = 0; r < 4; ++r)
#pragma unroll
        for (int c = 0; c < PHI_N; ++c) {
            float v = accv[r][c];
#pragma unroll
            for (int mask = 16; mask >= 1; mask >>= 1)
                v += __shfl_xor(v, mask, 32);
            accv[r][c] = v;
        }

    if (lane == 0) {
#pragma unroll
        for (int r = 0; r < 4; ++r)
#pragma unroll
            for (int c = 0; c < PHI_N; ++c)
                out[(size_t)(rowBase + r) * PHI_N + c] = accv[r][c];
    }
}

extern "C" void kernel_launch(void* const* d_in, const int* in_sizes, int n_in,
                              void* d_out, int out_size, void* d_ws, size_t ws_size,
                              hipStream_t stream) {
    const float* xs  = (const float*)d_in[0];   // [8192, 2048]
    const float* W   = (const float*)d_in[1];   // [1023, 2048]
    const float* b   = (const float*)d_in[2];   // [1023]
    const float* phi = (const float*)d_in[3];   // [1024, 24]
    float* out = (float*)d_out;                 // [8192, 24]

    // Workspace layout: probs (32MB) | Asw bf16 (32MB) | Bsw bf16 (4MB)
    char*   ws    = (char*)d_ws;
    float*  probs = (float*)(ws);
    __bf16* Asw   = (__bf16*)(ws + (size_t)32 * 1024 * 1024);
    __bf16* Bsw   = (__bf16*)(ws + (size_t)64 * 1024 * 1024);

    // Pre-swizzle to fragment-image bf16.
    sdt_swizzle_A<<<dim3(BATCH * (INPUT_DIM / 8) / 256), dim3(256), 0, stream>>>(xs, Asw);
    sdt_swizzle_B<<<dim3(N_PAD * (INPUT_DIM / 8) / 256), dim3(256), 0, stream>>>(W, Bsw);
    // WMMA GEMM + fused bias/sigmoid: 64 M-tiles x 8 N-tiles.
    sdt_gemm_sigmoid<<<dim3(64 * 8), dim3(256), 0, stream>>>(Asw, Bsw, b, probs);
    // Tree expansion + pp @ phi.
    sdt_tree_phi<<<dim3(BATCH / 32), dim3(256), 0, stream>>>(probs, phi, out);
}
